// StencilNet_38173669327080
// MI455X (gfx1250) — matrix operand order repaired and checked by reading, not compiled, not run on hardware
//
#include <hip/hip_runtime.h>
#include <hip/hip_bf16.h>
#include <stdint.h>

typedef __attribute__((ext_vector_type(16))) _Float16 v16h;
typedef __attribute__((ext_vector_type(8)))  _Float16 v8h;
typedef __attribute__((ext_vector_type(8)))  float    v8f;

#define HDIM 256
#define WDIM 256

// ---------------------------------------------------------------------------
// CDNA5 async global->LDS copy (16B per lane), tracked by ASYNCcnt.
// lds_off is the workgroup-relative LDS byte offset (low 32 bits of a generic
// pointer to __shared__), gaddr a 64-bit global address.
// ---------------------------------------------------------------------------
__device__ __forceinline__ void async_copy_b128(uint32_t lds_off, const void* g)
{
    asm volatile("global_load_async_to_lds_b128 %0, %1, off"
                 :
                 : "v"(lds_off), "v"((uint64_t)(uintptr_t)g)
                 : "memory");
}
__device__ __forceinline__ void wait_async_zero()
{
    asm volatile("s_wait_asynccnt 0" ::: "memory");
}

// ---------------------------------------------------------------------------
// Pack [B,H,W] C/vx/vy (f32) -> NHWC f16 activation with Cin padded to 32.
// ---------------------------------------------------------------------------
__global__ __launch_bounds__(256) void pack_input_kernel(
    const float* __restrict__ C, const float* __restrict__ vx,
    const float* __restrict__ vy, _Float16* __restrict__ act, int npix)
{
    int i = blockIdx.x * 256 + threadIdx.x;
    if (i >= npix) return;
    _Float16 tmp[32];
#pragma unroll
    for (int c = 0; c < 32; ++c) tmp[c] = (_Float16)0.0f;
    tmp[0] = (_Float16)((C[i] - 0.0f) / 1.0f);   // (C - SHIFT) / VAR
    tmp[1] = (_Float16)vx[i];
    tmp[2] = (_Float16)vy[i];
    _Float16* dst = act + (size_t)i * 32;
#pragma unroll
    for (int c = 0; c < 4; ++c) ((v8h*)dst)[c] = ((const v8h*)tmp)[c];
}

// ---------------------------------------------------------------------------
// Repack HWIO f32 weights into WMMA B-fragment order (f16):
//   wp[((tap*KS + ks)*NT + nt)*512 + lane*16 + h]
// lanes 0-15 : N = nt*16+lane,    K = ks*32 + h       (h = 0..15)
// lanes 16-31: N = nt*16+lane-16, K = ks*32 + 16 + h
// K >= KACT (layer-0 channel padding) -> 0.
// ---------------------------------------------------------------------------
__global__ __launch_bounds__(256) void repack_w_kernel(
    const float* __restrict__ w, _Float16* __restrict__ wp,
    int KPAD, int KACT, int COUT, int total)
{
    int o = blockIdx.x * 256 + threadIdx.x;
    if (o >= total) return;
    int hf   = o & 15;
    int lane = (o >> 4) & 31;
    int rest = o >> 9;
    int NT = COUT >> 4;
    int nt = rest % NT; rest /= NT;
    int KS = KPAD >> 5;
    int ks = rest % KS; rest /= KS;
    int tap = rest;                       // 0..8
    int K = ks * 32 + ((lane >= 16) ? 16 : 0) + hf;
    int N = nt * 16 + (lane & 15);
    float val = (K < KACT) ? w[((size_t)tap * KACT + K) * COUT + N] : 0.0f;
    wp[o] = (_Float16)val;
}

// ---------------------------------------------------------------------------
// Implicit-GEMM 3x3 periodic conv via v_wmma_f32_16x16x32_f16.
// Block = 256 threads (8 waves), tile = 16x16 pixels, halo 18x18xCIN in LDS
// (filled with global_load_async_to_lds_b128). Wave w computes tile rows
// 2w, 2w+1 (two M-tiles of 16 pixels) x COUT. Output staged through LDS for
// coalesced b128 stores.
// ---------------------------------------------------------------------------
template <int CIN, int COUT, bool RELU, bool F16OUT>
__global__ __launch_bounds__(256) void conv3x3_wmma_kernel(
    const _Float16* __restrict__ in, const _Float16* __restrict__ wp,
    const float* __restrict__ bias, void* __restrict__ outv)
{
    constexpr int KS   = CIN / 32;
    constexpr int NT   = COUT / 16;
    constexpr int HALO = 18;
    constexpr int ESZ  = F16OUT ? 2 : 4;
    constexpr int HALO_BYTES = HALO * HALO * CIN * 2;
    constexpr int OUT_BYTES  = 256 * COUT * ESZ;
    constexpr int SH_BYTES   = (HALO_BYTES > OUT_BYTES) ? HALO_BYTES : OUT_BYTES;
    __shared__ __align__(16) char smem[SH_BYTES];
    _Float16* halo = (_Float16*)smem;

    const int tilesPerImg = (HDIM / 16) * (WDIM / 16);
    const int b  = blockIdx.x / tilesPerImg;
    const int t  = blockIdx.x % tilesPerImg;
    const int h0 = (t / (WDIM / 16)) * 16;
    const int w0 = (t % (WDIM / 16)) * 16;

    __builtin_prefetch(wp, 0, 0);   // global_prefetch_b8 on the weight stream

    // Cooperative halo fill with periodic wrap: async global->LDS, 16B chunks.
    for (int p = threadIdx.x; p < HALO * HALO; p += 256) {
        const int i  = p / HALO, j = p % HALO;
        const int gh = (h0 + i - 1 + HDIM) % HDIM;
        const int gw = (w0 + j - 1 + WDIM) % WDIM;
        const char* src =
            (const char*)(in + (((size_t)b * HDIM + gh) * WDIM + gw) * CIN);
        const uint32_t dlds = (uint32_t)(uintptr_t)(halo + (size_t)p * CIN);
#pragma unroll
        for (int c = 0; c < CIN / 8; ++c)
            async_copy_b128(dlds + c * 16, src + c * 16);
    }
    wait_async_zero();
    __syncthreads();

    const int wave = threadIdx.x >> 5;
    const int lane = threadIdx.x & 31;
    const int lm   = lane & 15;
    const int hiK  = (lane >> 4) & 1;   // 1 for lanes 16-31
    const int r0   = wave * 2;

    v8f zeroAcc = {};
    v8f acc[2][NT];
#pragma unroll
    for (int rr = 0; rr < 2; ++rr)
#pragma unroll
        for (int nt = 0; nt < NT; ++nt) acc[rr][nt] = zeroAcc;

    for (int ks = 0; ks < KS; ++ks) {
        for (int tap = 0; tap < 9; ++tap) {
            const int ty = tap / 3, tx = tap % 3;
            // A fragment (16x32 f16): lanes 0-15 hold K {0..15},
            // lanes 16-31 hold K {16..31} of this 32-wide K step.
            union { v16h v; v8h h[2]; } a0, a1;
            {
                const _Float16* s0 =
                    halo + ((size_t)(r0 + ty) * HALO + (lm + tx)) * CIN + ks * 32 + hiK * 16;
                a0.h[0] = *(const v8h*)(s0);
                a0.h[1] = *(const v8h*)(s0 + 8);
                const _Float16* s1 =
                    halo + ((size_t)(r0 + 1 + ty) * HALO + (lm + tx)) * CIN + ks * 32 + hiK * 16;
                a1.h[0] = *(const v8h*)(s1);
                a1.h[1] = *(const v8h*)(s1 + 8);
            }
#pragma unroll
            for (int nt = 0; nt < NT; ++nt) {
                union { v16h v; v8h h[2]; } bf;
                const _Float16* sw =
                    wp + (((size_t)(tap * KS + ks) * NT + nt) << 9) + lane * 16;
                bf.h[0] = *(const v8h*)(sw);
                bf.h[1] = *(const v8h*)(sw + 8);
                acc[0][nt] = __builtin_amdgcn_wmma_f32_16x16x32_f16(
                    false, a0.v, false, bf.v, (short)0, acc[0][nt], false, false);
                acc[1][nt] = __builtin_amdgcn_wmma_f32_16x16x32_f16(
                    false, a1.v, false, bf.v, (short)0, acc[1][nt], false, false);
            }
        }
    }

    // All halo reads are done; reuse LDS to stage the output tile NHWC so the
    // global stores below are fully coalesced b128s.
    __syncthreads();
#pragma unroll
    for (int rr = 0; rr < 2; ++rr) {
#pragma unroll
        for (int nt = 0; nt < NT; ++nt) {
            const int n = nt * 16 + lm;
            const float bv = bias[n];
#pragma unroll
            for (int q = 0; q < 8; ++q) {
                const int m = q + hiK * 8;           // D layout: lane<16 M=q, else M=q+8
                const int p = (r0 + rr) * 16 + m;    // pixel within tile
                float v = acc[rr][nt][q] + bv;
                if (RELU) v = fmaxf(v, 0.0f);
                if (F16OUT) ((_Float16*)smem)[p * COUT + n] = (_Float16)v;
                else        ((float*)smem)[p * COUT + n]    = v;
            }
        }
    }
    __syncthreads();

    // Cooperative coalesced copy-out: each tile row is contiguous in NHWC.
    constexpr int ROW_BYTES = 16 * COUT * ESZ;
    constexpr int NCHUNK    = 16 * (ROW_BYTES / 16);
    for (int c = threadIdx.x; c < NCHUNK; c += 256) {
        const int r = c / (ROW_BYTES / 16);
        const int o = c % (ROW_BYTES / 16);
        const v8h val = *(const v8h*)(smem + (size_t)r * ROW_BYTES + o * 16);
        char* dst = (char*)outv +
            ((((size_t)b * HDIM + (h0 + r)) * WDIM + w0) * COUT) * ESZ + (size_t)o * 16;
        *(v8h*)dst = val;
    }
}

// ---------------------------------------------------------------------------
// Final 4x4 periodic stencil contraction: out = sum_s coeffs[...,s]*roll(C).
// ---------------------------------------------------------------------------
__global__ __launch_bounds__(256) void stencil_apply_kernel(
    const float* __restrict__ coeffs, const float* __restrict__ C,
    float* __restrict__ out, int npix)
{
    int idx = blockIdx.x * 256 + threadIdx.x;
    if (idx >= npix) return;
    int w = idx % WDIM;
    int h = (idx / WDIM) % HDIM;
    int b = idx / (WDIM * HDIM);
    const float* cf = coeffs + (size_t)idx * 16;
    float s = 0.0f;
#pragma unroll
    for (int i = 0; i < 4; ++i) {
        const int hh = (h + i - 2 + HDIM) % HDIM;
#pragma unroll
        for (int j = 0; j < 4; ++j) {
            const int ww = (w + j - 2 + WDIM) % WDIM;
            s += cf[i * 4 + j] * C[((size_t)b * HDIM + hh) * WDIM + ww];
        }
    }
    out[idx] = s;
}

// ---------------------------------------------------------------------------
extern "C" void kernel_launch(void* const* d_in, const int* in_sizes, int n_in,
                              void* d_out, int out_size, void* d_ws, size_t ws_size,
                              hipStream_t stream)
{
    const float* C    = (const float*)d_in[0];
    const float* vx   = (const float*)d_in[1];
    const float* vy   = (const float*)d_in[2];
    const float* w0   = (const float*)d_in[3];
    const float* b0   = (const float*)d_in[4];
    const float* w1   = (const float*)d_in[5];
    const float* b1   = (const float*)d_in[6];
    const float* w2   = (const float*)d_in[7];
    const float* b2   = (const float*)d_in[8];
    const float* w3   = (const float*)d_in[9];
    const float* b3   = (const float*)d_in[10];
    const float* w4   = (const float*)d_in[11];
    const float* b4   = (const float*)d_in[12];
    const float* wout = (const float*)d_in[13];
    const float* bout = (const float*)d_in[14];
    float* out = (float*)d_out;

    const int Bn = in_sizes[0] / (HDIM * WDIM);      // 8
    const size_t npix = (size_t)Bn * HDIM * WDIM;    // 524288

    // Workspace layout
    char* ws = (char*)d_ws;
    _Float16* actA   = (_Float16*)ws;                          // npix*64 f16 (64MB)
    _Float16* actB   = (_Float16*)(ws + npix * 64 * 2);        // npix*64 f16 (64MB)
    float*    coeffs = (float*)(ws + 2 * npix * 64 * 2);       // npix*16 f32 (32MB)
    _Float16* wp0 = (_Float16*)(ws + 2 * npix * 64 * 2 + npix * 16 * 4);
    _Float16* wp1 = wp0 + 9 * 32 * 64;
    _Float16* wp2 = wp1 + 9 * 64 * 64;
    _Float16* wp3 = wp2 + 9 * 64 * 64;
    _Float16* wp4 = wp3 + 9 * 64 * 64;
    _Float16* wpo = wp4 + 9 * 64 * 64;

    // 1) pack input (Cin padded 3->32)
    pack_input_kernel<<<(int)((npix + 255) / 256), 256, 0, stream>>>(C, vx, vy, actA, (int)npix);

    // 2) repack weights into WMMA B-fragment layout
    repack_w_kernel<<<(9 * 32 * 64 + 255) / 256, 256, 0, stream>>>(w0, wp0, 32, 3, 64, 9 * 32 * 64);
    repack_w_kernel<<<(9 * 64 * 64 + 255) / 256, 256, 0, stream>>>(w1, wp1, 64, 64, 64, 9 * 64 * 64);
    repack_w_kernel<<<(9 * 64 * 64 + 255) / 256, 256, 0, stream>>>(w2, wp2, 64, 64, 64, 9 * 64 * 64);
    repack_w_kernel<<<(9 * 64 * 64 + 255) / 256, 256, 0, stream>>>(w3, wp3, 64, 64, 64, 9 * 64 * 64);
    repack_w_kernel<<<(9 * 64 * 64 + 255) / 256, 256, 0, stream>>>(w4, wp4, 64, 64, 64, 9 * 64 * 64);
    repack_w_kernel<<<(9 * 64 * 16 + 255) / 256, 256, 0, stream>>>(wout, wpo, 64, 64, 16, 9 * 64 * 16);

    // 3) conv stack (implicit-GEMM WMMA), 16x16 pixel tiles
    const int convBlocks = Bn * (HDIM / 16) * (WDIM / 16);
    conv3x3_wmma_kernel<32, 64, true,  true ><<<convBlocks, 256, 0, stream>>>(actA, wp0, b0, actB);
    conv3x3_wmma_kernel<64, 64, true,  true ><<<convBlocks, 256, 0, stream>>>(actB, wp1, b1, actA);
    conv3x3_wmma_kernel<64, 64, true,  true ><<<convBlocks, 256, 0, stream>>>(actA, wp2, b2, actB);
    conv3x3_wmma_kernel<64, 64, true,  true ><<<convBlocks, 256, 0, stream>>>(actB, wp3, b3, actA);
    conv3x3_wmma_kernel<64, 64, true,  true ><<<convBlocks, 256, 0, stream>>>(actA, wp4, b4, actB);
    conv3x3_wmma_kernel<64, 16, false, false><<<convBlocks, 256, 0, stream>>>(actB, wpo, bout, coeffs);

    // 4) stencil contraction
    stencil_apply_kernel<<<(int)((npix + 255) / 256), 256, 0, stream>>>(coeffs, C, out, (int)npix);
}
